// Model_53154515255334
// MI455X (gfx1250) — compile-verified, matrix operand order checked by reading, not run on hardware
//
#include <hip/hip_runtime.h>
#include <hip/hip_bf16.h>

// ---------------------------------------------------------------------------
// Model dims
// ---------------------------------------------------------------------------
#define Bn    256
#define Ln    211
#define Vn    32
#define EVn   16
#define PROJ  512
#define BASEn 64
#define BASEE 128
#define H1n   256
#define H2n   128
#define Dn    (PROJ + BASEE)          // 640
#define K1    1024                    // fused [xe|te] K
#define MROWS (Bn * Ln)               // 54016  (divisible by 128)
#define MT1   (MROWS / 16)            // 3376 A mfrags
#define LT2   14                      // ceil(211/16) mfrags per batch
#define KT2   (PROJ / 32)             // 16 k-tiles for GEMM2

typedef __attribute__((ext_vector_type(16))) _Float16 v16h;
typedef __attribute__((ext_vector_type(8)))  float    v8f;

static __device__ __forceinline__ v8f wmma_f16(v16h a, v16h b, v8f c) {
  return __builtin_amdgcn_wmma_f32_16x16x32_f16(
      /*neg_a=*/false, a, /*neg_b=*/false, b,
      /*c_mod=*/(short)0, c, /*reuse_a=*/false, /*reuse_b=*/false);
}

// CDNA5 A/B fragment element mapping (16x32 f16 per ISA 7.12.2):
//   lane = (row%16) + 16*h,  h = (k>>3)&1,  j = (k&7) | ((k&16)>>1)
static __device__ __forceinline__ int frag_h(int k) { return (k >> 3) & 1; }
static __device__ __forceinline__ int frag_j(int k) { return (k & 7) | ((k & 16) >> 1); }

// Ex stored as packed A-fragments per batch: [b][lt][kt][lane*16+j]
static __device__ __forceinline__ size_t exIdx(int b, int l, int p) {
  int lt = l >> 4, m = l & 15;
  int kt = p >> 5, k = p & 31;
  int lane = m + (frag_h(k) << 4);
  return (size_t)b * (LT2 * KT2 * 512) + (size_t)((lt * KT2 + kt) * 512) + lane * 16 + frag_j(k);
}

// ---------------------------------------------------------------------------
// Workspace layout (bytes)
// ---------------------------------------------------------------------------
#define SZ_APACK  ((size_t)MROWS * K1 * 2)              // 110.6 MB  f16 A frags
#define SZ_W1P    ((size_t)K1 * PROJ * 2)               // 1 MB      f16 B frags
#define SZ_EXPACK ((size_t)Bn * LT2 * 16 * PROJ * 2)    // 58.7 MB   f16 Ex frags (padded)
#define SZ_EXM    ((size_t)Bn * PROJ * 4)
#define SZ_EFV    ((size_t)PROJ * 4)
#define SZ_WK     ((size_t)PROJ * 3 * 4)
#define SZ_MSM    ((size_t)Bn * PROJ * 4)
#define SZ_INVZ   ((size_t)Bn * PROJ * 4)
#define SZ_EXOUT  ((size_t)Bn * Ln * PROJ * 4)          // 110.6 MB
#define SZ_EXALL  ((size_t)Bn * Dn * 4)
#define SZ_ADJ    ((size_t)Bn * Bn * 4)
#define SZ_T1     ((size_t)Bn * H1n * 4)
#define SZ_H      ((size_t)Bn * H1n * 4)
#define SZ_T2     ((size_t)Bn * H2n * 4)
#define SZ_XOTH   ((size_t)Bn * H2n * 4)
#define SZ_XO     ((size_t)Bn * Dn * 4)

#define OFF_APACK  ((size_t)0)
#define OFF_W1P    (OFF_APACK  + SZ_APACK)
#define OFF_EXPACK (OFF_W1P    + SZ_W1P)
#define OFF_EXM    (OFF_EXPACK + SZ_EXPACK)
#define OFF_EFV    (OFF_EXM    + SZ_EXM)
#define OFF_WK     (OFF_EFV    + SZ_EFV)
#define OFF_MSM    (OFF_WK     + SZ_WK)
#define OFF_INVZ   (OFF_MSM    + SZ_MSM)
#define OFF_EXOUT  (OFF_INVZ   + SZ_INVZ)
#define OFF_EXALL  (OFF_EXOUT  + SZ_EXOUT)
#define OFF_ADJ    (OFF_EXALL  + SZ_EXALL)
#define OFF_T1     (OFF_ADJ    + SZ_ADJ)
#define OFF_H      (OFF_T1     + SZ_T1)
#define OFF_T2     (OFF_H      + SZ_H)
#define OFF_XOTH   (OFF_T2     + SZ_T2)
#define OFF_XO     (OFF_XOTH   + SZ_XOTH)

// ---------------------------------------------------------------------------
// 1) Embedding: per-feature 1->4->16 tanh MLPs for value & time branches,
//    written directly as packed f16 A-fragments ([xe|te], K=1024).
// ---------------------------------------------------------------------------
__global__ __launch_bounds__(256) void k_embed(
    const float* __restrict__ x, const float* __restrict__ tim,
    const float* __restrict__ Wv1, const float* __restrict__ bv1, const float* __restrict__ Wv2,
    const float* __restrict__ Wt1, const float* __restrict__ bt1, const float* __restrict__ Wt2,
    _Float16* __restrict__ A) {
  int gid = blockIdx.x * 256 + threadIdx.x;         // MROWS*64 units
  if (gid >= MROWS * 64) return;
  int u = gid & 63, row = gid >> 6;
  int v = u & 31, br = u >> 5;                      // br: 0=value, 1=time
  float s = br ? tim[row] : x[row * Vn + v];
  const float* W1 = br ? Wt1 : Wv1;
  const float* B1 = br ? bt1 : bv1;
  const float* W2 = br ? Wt2 : Wv2;
  float h0 = tanhf(s * W1[v * 4 + 0] + B1[v * 4 + 0]);
  float h1 = tanhf(s * W1[v * 4 + 1] + B1[v * 4 + 1]);
  float h2 = tanhf(s * W1[v * 4 + 2] + B1[v * 4 + 2]);
  float h3 = tanhf(s * W1[v * 4 + 3] + B1[v * 4 + 3]);
  int mt = row >> 4, m = row & 15;
#pragma unroll
  for (int d = 0; d < 16; ++d) {
    float o = h0 * W2[(v * 4 + 0) * 16 + d] + h1 * W2[(v * 4 + 1) * 16 + d] +
              h2 * W2[(v * 4 + 2) * 16 + d] + h3 * W2[(v * 4 + 3) * 16 + d];
    int kg = br * 512 + v * 16 + d;
    int kt = kg >> 5, k = kg & 31;
    A[(size_t)(mt * 32 + kt) * 512 + (m + (frag_h(k) << 4)) * 16 + frag_j(k)] = (_Float16)o;
  }
}

// 2) Pack fused weight [Wx_w ; Wtp_w] (1024x512 f32) into f16 B-fragments.
__global__ __launch_bounds__(256) void k_packW(
    const float* __restrict__ Wx, const float* __restrict__ Wtp, _Float16* __restrict__ W) {
  int idx = blockIdx.x * 256 + threadIdx.x;         // K1*PROJ
  if (idx >= K1 * PROJ) return;
  int n = idx & 511, k = idx >> 9;
  float val = (k < 512) ? Wx[k * 512 + n] : Wtp[(k - 512) * 512 + n];
  int kt = k >> 5, kk = k & 31, nt = n >> 4;
  W[(size_t)(kt * 32 + nt) * 512 + ((n & 15) + (frag_h(kk) << 4)) * 16 + frag_j(kk)] = (_Float16)val;
}

// 3) Efv = flatten((emb_f[f_idx].reshape(V,40)) @ fea_w + fea_b)  -> [512]
__global__ __launch_bounds__(512) void k_efv(
    const int* __restrict__ fidx, const float* __restrict__ emb_f,
    const float* __restrict__ fea_w, const float* __restrict__ fea_b,
    float* __restrict__ Efv) {
  int t = threadIdx.x;                              // 512
  int v = t >> 4, d = t & 15;
  float acc = fea_b[d];
  for (int i = 0; i < 5; ++i) {
    int fi = fidx[v * 5 + i];
    for (int e = 0; e < 8; ++e)
      acc += emb_f[fi * 8 + e] * fea_w[(i * 8 + e) * 16 + d];
  }
  Efv[t] = acc;
}

// 4) Wk[c,k] = (1/sqrt(V)) * sum_i conv_w[c,i,k] * Efv[i]
__global__ __launch_bounds__(256) void k_wk(
    const float* __restrict__ conv_w, const float* __restrict__ Efv, float* __restrict__ Wk) {
  int idx = blockIdx.x * 256 + threadIdx.x;         // 1536
  if (idx >= PROJ * 3) return;
  int c = idx / 3, k = idx % 3;
  float acc = 0.f;
  for (int i = 0; i < PROJ; ++i) acc += conv_w[(c * PROJ + i) * 3 + k] * Efv[i];
  Wk[idx] = acc * 0.17677669529663687f;             // 1/sqrt(32)
}

// ---------------------------------------------------------------------------
// 5) GEMM-1 (WMMA f16): Ex = A[54016,1024] @ W[1024,512] + (Wx_b + Wtp_b)
//    Fragments pre-packed -> direct global v16h loads, no LDS.
//    WG = 128(M) x 64(N); 8 waves in 4x2 grid, each wave 2x2 frags.
//    Per-kt strides are compile-time constants -> pointer increments so the
//    backend folds them into load immediate offsets.
// ---------------------------------------------------------------------------
__global__ __launch_bounds__(256) void k_gemm1(
    const _Float16* __restrict__ A, const _Float16* __restrict__ W,
    const float* __restrict__ Wx_b, const float* __restrict__ Wtp_b,
    _Float16* __restrict__ ExPack) {
  int wid = threadIdx.x >> 5, lane = threadIdx.x & 31;
  int wm = wid & 3, wn = wid >> 2;
  int mf0 = blockIdx.x * 8 + wm * 2;                // global mfrag
  int nf0 = blockIdx.y * 4 + wn * 2;                // global nfrag
  const _Float16* pa0 = A + (size_t)mf0 * (32 * 512) + lane * 16;
  const _Float16* pa1 = pa0 + 32 * 512;
  const _Float16* pb0 = W + (size_t)nf0 * 512 + lane * 16;
  const _Float16* pb1 = pb0 + 512;
  v8f acc00 = {}, acc01 = {}, acc10 = {}, acc11 = {};
#pragma unroll 4
  for (int kt = 0; kt < 32; ++kt) {
    v16h a0 = *(const v16h*)pa0;
    v16h a1 = *(const v16h*)pa1;
    v16h b0 = *(const v16h*)pb0;
    v16h b1 = *(const v16h*)pb1;
    acc00 = wmma_f16(a0, b0, acc00);
    acc01 = wmma_f16(a0, b1, acc01);
    acc10 = wmma_f16(a1, b0, acc10);
    acc11 = wmma_f16(a1, b1, acc11);
    pa0 += 512;           // next kt within same mfrag
    pa1 += 512;
    pb0 += 32 * 512;      // next kt row of B frags
    pb1 += 32 * 512;
  }
  int nl = lane & 15, mh = (lane >> 4) * 8;
#pragma unroll
  for (int i = 0; i < 2; ++i)
#pragma unroll
    for (int jf = 0; jf < 2; ++jf) {
      v8f acc = (i == 0) ? (jf == 0 ? acc00 : acc01) : (jf == 0 ? acc10 : acc11);
      int mfg = mf0 + i, nfg = nf0 + jf;
#pragma unroll
      for (int r = 0; r < 8; ++r) {
        int mg = mfg * 16 + mh + r;
        int b = mg / Ln, l = mg % Ln;
        int n = nfg * 16 + nl;
        float vv = acc[r] + Wx_b[n] + Wtp_b[n];
        ExPack[exIdx(b, l, n)] = (_Float16)vv;
      }
    }
}

// 6) Exm[b,p] = sum_l vseq_w[l]*Ex[b,l,p] + vseq_b
__global__ __launch_bounds__(512) void k_exm(
    const _Float16* __restrict__ ExPack, const float* __restrict__ vw,
    const float* __restrict__ vb, float* __restrict__ Exm) {
  __shared__ float sw[Ln];
  int b = blockIdx.x, tid = threadIdx.x;
  for (int l = tid; l < Ln; l += 512) sw[l] = vw[l];
  __syncthreads();
  int p = tid;
  float acc = vb[0];
  for (int l = 0; l < Ln; ++l) acc += sw[l] * (float)ExPack[exIdx(b, l, p)];
  Exm[b * PROJ + p] = acc;
}

// 7) Softmax stats over p for Ec[b,c,p] = Wk0*Exm[p-1]+Wk1*Exm[p]+Wk2*Exm[p+1]+cb[c]
__global__ __launch_bounds__(256) void k_stats(
    const float* __restrict__ Exm, const float* __restrict__ Wk,
    const float* __restrict__ cb, float* __restrict__ Msm, float* __restrict__ invZ) {
  __shared__ float sx[PROJ + 2];
  int b = blockIdx.x, tid = threadIdx.x;
  for (int p = tid; p < PROJ; p += 256) sx[p + 1] = Exm[b * PROJ + p];
  if (tid == 0) { sx[0] = 0.f; sx[PROJ + 1] = 0.f; }
  __syncthreads();
  for (int c = tid; c < PROJ; c += 256) {
    float w0 = Wk[c * 3 + 0], w1 = Wk[c * 3 + 1], w2 = Wk[c * 3 + 2], bb = cb[c];
    float mx = -3.4e38f;
    for (int p = 0; p < PROJ; ++p) {
      float e = w0 * sx[p] + w1 * sx[p + 1] + w2 * sx[p + 2] + bb;
      mx = fmaxf(mx, e);
    }
    float sum = 0.f;
    for (int p = 0; p < PROJ; ++p) {
      float e = w0 * sx[p] + w1 * sx[p + 1] + w2 * sx[p + 2] + bb;
      sum += __expf(e - mx);
    }
    Msm[b * PROJ + c] = mx;
    invZ[b * PROJ + c] = 1.f / sum;
  }
}

// ---------------------------------------------------------------------------
// 8) GEMM-2 (WMMA f16): Ex_out[b,l,c] = sum_p att[b,c,p] * Ex[b,l,p]
//    att regenerated on the fly into a 64 KB LDS fragment tile (64 c x 512 p).
//    WG per (b, c-tile of 64); 8 waves: wave = (c-subfrag, half of mfrags).
// ---------------------------------------------------------------------------
__global__ __launch_bounds__(256) void k_gemm2(
    const _Float16* __restrict__ ExPack, const float* __restrict__ Exm,
    const float* __restrict__ Wk, const float* __restrict__ cb,
    const float* __restrict__ Msm, const float* __restrict__ invZ,
    float* __restrict__ ExOut) {
  __shared__ __align__(16) _Float16 attF[64 * 512];       // 64 frags x 512 halfs = 64 KB
  int b = blockIdx.x, ct = blockIdx.y;                    // ct in [0,8): c0 = ct*64
  int tid = threadIdx.x;
  const float* exm = Exm + b * PROJ;
  // phase 1: build att B-fragments (B[k=p][n=c])
  for (int idx = tid; idx < 64 * 512; idx += 256) {
    int p = idx >> 6, cl = idx & 63;
    int c = ct * 64 + cl;
    float xm1 = (p > 0)   ? exm[p - 1] : 0.f;
    float x0  = exm[p];
    float xp1 = (p < 511) ? exm[p + 1] : 0.f;
    float e = Wk[c * 3 + 0] * xm1 + Wk[c * 3 + 1] * x0 + Wk[c * 3 + 2] * xp1 + cb[c];
    float a = __expf(e - Msm[b * PROJ + c]) * invZ[b * PROJ + c];
    int k = p & 31;
    int frag = (p >> 5) * 4 + (cl >> 4);
    attF[frag * 512 + ((cl & 15) + (frag_h(k) << 4)) * 16 + frag_j(k)] = (_Float16)a;
  }
  __syncthreads();
  // phase 2: 14 mfrags x 16 kt WMMA
  int wid = tid >> 5, lane = tid & 31;
  int nf = wid & 3;                                       // c sub-tile
  int mhalf = wid >> 2;                                   // mfrags [0..6] or [7..13]
  const _Float16* exb = ExPack + (size_t)b * (LT2 * KT2 * 512) + lane * 16;
  const _Float16* lbase = &attF[nf * 512 + lane * 16];
  int nl = lane & 15, mh = (lane >> 4) * 8;
  for (int mf = mhalf * 7; mf < mhalf * 7 + 7; ++mf) {
    const _Float16* pa = exb + (size_t)mf * (KT2 * 512);
    const _Float16* pb = lbase;
    v8f acc = {};
#pragma unroll 4
    for (int kt = 0; kt < KT2; ++kt) {
      v16h a  = *(const v16h*)pa;
      v16h bf = *(const v16h*)pb;
      acc = wmma_f16(a, bf, acc);
      pa += 512;            // next kt frag of Ex
      pb += 4 * 512;        // next kt row of att frags
    }
#pragma unroll
    for (int r = 0; r < 8; ++r) {
      int l = mf * 16 + mh + r;
      if (l < Ln) {
        int c = ct * 64 + nf * 16 + nl;
        ExOut[((size_t)b * Ln + l) * PROJ + c] = acc[r];
      }
    }
  }
}

// 9) Visit attention + Exs + ebase -> Ex_all[b, 640]
__global__ __launch_bounds__(256) void k_visit(
    const float* __restrict__ ExOut, const float* __restrict__ visit_w,
    const float* __restrict__ visit_b, const float* __restrict__ xbase,
    const float* __restrict__ sproj_w, const float* __restrict__ sproj_b,
    float* __restrict__ ExAll) {
  __shared__ float ss[Ln];
  int b = blockIdx.x, tid = threadIdx.x;
  if (tid < Ln) {
    const float* row = ExOut + ((size_t)b * Ln + tid) * PROJ;
    float acc = visit_b[0];
    for (int p = 0; p < PROJ; ++p) acc += row[p] * visit_w[p];
    ss[tid] = acc;
  }
  __syncthreads();
  if (tid == 0) {
    float mx = -3.4e38f;
    for (int l = 0; l < Ln; ++l) mx = fmaxf(mx, ss[l]);
    float sum = 0.f;
    for (int l = 0; l < Ln; ++l) sum += __expf(ss[l] - mx);
    float inv = 1.f / sum;
    for (int l = 0; l < Ln; ++l) ss[l] = __expf(ss[l] - mx) * inv;
  }
  __syncthreads();
  for (int p = tid; p < PROJ; p += 256) {
    float acc = 0.f;
    for (int l = 0; l < Ln; ++l) acc += ss[l] * ExOut[((size_t)b * Ln + l) * PROJ + p];
    ExAll[b * Dn + p] = acc;
  }
  if (tid < BASEE) {
    float acc = sproj_b[tid];
    for (int k = 0; k < BASEn; ++k) acc += xbase[b * BASEn + k] * sproj_w[k * BASEE + tid];
    ExAll[b * Dn + PROJ + tid] = acc;
  }
}

// 10) adj[i,j] = t > phi ? t : 0, t = dot(ExAll[i],ExAll[j]) / PROJ^2
__global__ __launch_bounds__(256) void k_adj(
    const float* __restrict__ ExAll, const float* __restrict__ phi, float* __restrict__ adj) {
  __shared__ float ri[Dn];
  int i = blockIdx.x, j = threadIdx.x;
  for (int k = j; k < Dn; k += 256) ri[k] = ExAll[i * Dn + k];
  __syncthreads();
  float acc = 0.f;
  for (int k = 0; k < Dn; ++k) acc += ri[k] * ExAll[j * Dn + k];
  acc *= (1.f / ((float)PROJ * (float)PROJ));
  adj[i * Bn + j] = (acc > phi[0]) ? acc : 0.f;
}

// 11) generic f32 GEMM: C[row,n] = (bias?bias[n]:0) + A[row,:]@B[:,n], opt relu
__global__ __launch_bounds__(256) void k_mm(
    const float* __restrict__ A, const float* __restrict__ Bm,
    const float* __restrict__ bias, float* __restrict__ C,
    int K, int N, int relu) {
  extern __shared__ float arow[];
  int row = blockIdx.x;
  for (int k = threadIdx.x; k < K; k += 256) arow[k] = A[row * K + k];
  __syncthreads();
  int n = blockIdx.y * 256 + threadIdx.x;
  if (n >= N) return;
  float acc = bias ? bias[n] : 0.f;
  for (int k = 0; k < K; ++k) acc += arow[k] * Bm[k * N + n];
  if (relu) acc = fmaxf(acc, 0.f);
  C[row * N + n] = acc;
}

// 12) row-wise log_softmax [Bn, H2n]
__global__ __launch_bounds__(128) void k_lsm(const float* __restrict__ in, float* __restrict__ out) {
  __shared__ float red[128];
  int r = blockIdx.x, t = threadIdx.x;
  float v = in[r * H2n + t];
  red[t] = v; __syncthreads();
  for (int s = 64; s > 0; s >>= 1) { if (t < s) red[t] = fmaxf(red[t], red[t + s]); __syncthreads(); }
  float mx = red[0]; __syncthreads();
  red[t] = __expf(v - mx); __syncthreads();
  for (int s = 64; s > 0; s >>= 1) { if (t < s) red[t] += red[t + s]; __syncthreads(); }
  out[r * H2n + t] = v - mx - logf(red[0]);
}

// 13) gated fusion + final softmax; writes d_out = [probs(256x2) | x_final(256x640)]
__global__ __launch_bounds__(256) void k_final(
    const float* __restrict__ ExAll, const float* __restrict__ xo,
    const float* __restrict__ agg1_w, const float* __restrict__ agg1_b,
    const float* __restrict__ agg2_w, const float* __restrict__ agg2_b,
    const float* __restrict__ wy_w, const float* __restrict__ wy_b,
    float* __restrict__ out) {
  __shared__ float red[256];
  __shared__ float xfs[Dn];
  int b = blockIdx.x, t = threadIdx.x;
  // gamma
  float p0 = 0.f;
  for (int p = t; p < Dn; p += 256) p0 += ExAll[b * Dn + p] * agg1_w[p];
  red[t] = p0; __syncthreads();
  for (int s = 128; s > 0; s >>= 1) { if (t < s) red[t] += red[t + s]; __syncthreads(); }
  float gamma = 1.f / (1.f + __expf(-(red[0] + agg1_b[0]))); __syncthreads();
  // eta
  p0 = 0.f;
  for (int p = t; p < Dn; p += 256) p0 += xo[b * Dn + p] * agg2_w[p];
  red[t] = p0; __syncthreads();
  for (int s = 128; s > 0; s >>= 1) { if (t < s) red[t] += red[t + s]; __syncthreads(); }
  float eta = 1.f / (1.f + __expf(-(red[0] + agg2_b[0]))); __syncthreads();
  float g = gamma / (gamma + eta);
  for (int p = t; p < Dn; p += 256) {
    float xf = g * ExAll[b * Dn + p] + (1.f - g) * xo[b * Dn + p];
    xfs[p] = xf;
    out[Bn * 2 + b * Dn + p] = xf;                       // x_final
  }
  __syncthreads();
  float l0 = 0.f, l1 = 0.f;
  for (int p = t; p < Dn; p += 256) { l0 += xfs[p] * wy_w[p * 2 + 0]; l1 += xfs[p] * wy_w[p * 2 + 1]; }
  red[t] = l0; __syncthreads();
  for (int s = 128; s > 0; s >>= 1) { if (t < s) red[t] += red[t + s]; __syncthreads(); }
  l0 = red[0] + wy_b[0]; __syncthreads();
  red[t] = l1; __syncthreads();
  for (int s = 128; s > 0; s >>= 1) { if (t < s) red[t] += red[t + s]; __syncthreads(); }
  l1 = red[0] + wy_b[1];
  if (t == 0) {
    float mx = fmaxf(l0, l1);
    float e0 = __expf(l0 - mx), e1 = __expf(l1 - mx);
    float inv = 1.f / (e0 + e1);
    out[b * 2 + 0] = e0 * inv;
    out[b * 2 + 1] = e1 * inv;
  }
}

// ---------------------------------------------------------------------------
extern "C" void kernel_launch(void* const* d_in, const int* in_sizes, int n_in,
                              void* d_out, int out_size, void* d_ws, size_t ws_size,
                              hipStream_t stream) {
  (void)in_sizes; (void)n_in; (void)out_size; (void)ws_size;
  const int*   f_idx  = (const int*)  d_in[0];
  const float* x      = (const float*)d_in[1];
  const float* tim    = (const float*)d_in[2];
  const float* xbase  = (const float*)d_in[3];
  /* d_in[4] = mask, unused by reference */
  const float* Wv1    = (const float*)d_in[5];
  const float* bv1    = (const float*)d_in[6];
  const float* Wv2    = (const float*)d_in[7];
  const float* Wt1    = (const float*)d_in[8];
  const float* bt1    = (const float*)d_in[9];
  const float* Wt2    = (const float*)d_in[10];
  const float* Wx_w   = (const float*)d_in[11];
  const float* Wx_b   = (const float*)d_in[12];
  const float* Wtp_w  = (const float*)d_in[13];
  const float* Wtp_b  = (const float*)d_in[14];
  const float* emb_f  = (const float*)d_in[15];
  const float* fea_w  = (const float*)d_in[16];
  const float* fea_b  = (const float*)d_in[17];
  const float* vseq_w = (const float*)d_in[18];
  const float* vseq_b = (const float*)d_in[19];
  const float* conv_w = (const float*)d_in[20];
  const float* conv_b = (const float*)d_in[21];
  const float* visit_w= (const float*)d_in[22];
  const float* visit_b= (const float*)d_in[23];
  const float* sproj_w= (const float*)d_in[24];
  const float* sproj_b= (const float*)d_in[25];
  const float* phi    = (const float*)d_in[26];
  const float* gc1_w  = (const float*)d_in[27];
  const float* gc1_b  = (const float*)d_in[28];
  const float* gc2_w  = (const float*)d_in[29];
  const float* gc2_b  = (const float*)d_in[30];
  const float* agg_w  = (const float*)d_in[31];
  const float* agg_b  = (const float*)d_in[32];
  const float* agg1_w = (const float*)d_in[33];
  const float* agg1_b = (const float*)d_in[34];
  const float* agg2_w = (const float*)d_in[35];
  const float* agg2_b = (const float*)d_in[36];
  const float* wy_w   = (const float*)d_in[37];
  const float* wy_b   = (const float*)d_in[38];

  char* ws = (char*)d_ws;
  _Float16* Apack  = (_Float16*)(ws + OFF_APACK);
  _Float16* W1p    = (_Float16*)(ws + OFF_W1P);
  _Float16* ExPack = (_Float16*)(ws + OFF_EXPACK);
  float* Exm   = (float*)(ws + OFF_EXM);
  float* Efv   = (float*)(ws + OFF_EFV);
  float* Wk    = (float*)(ws + OFF_WK);
  float* Msm   = (float*)(ws + OFF_MSM);
  float* invZ  = (float*)(ws + OFF_INVZ);
  float* ExOut = (float*)(ws + OFF_EXOUT);
  float* ExAll = (float*)(ws + OFF_EXALL);
  float* adj   = (float*)(ws + OFF_ADJ);
  float* t1    = (float*)(ws + OFF_T1);
  float* hbuf  = (float*)(ws + OFF_H);
  float* t2    = (float*)(ws + OFF_T2);
  float* xoth  = (float*)(ws + OFF_XOTH);
  float* xo    = (float*)(ws + OFF_XO);
  float* outp  = (float*)d_out;

  // pad rows of ExPack (l >= 211) must be zero for GEMM-2
  hipMemsetAsync(ExPack, 0, SZ_EXPACK, stream);

  k_embed<<<dim3((MROWS * 64) / 256), 256, 0, stream>>>(x, tim, Wv1, bv1, Wv2, Wt1, bt1, Wt2, Apack);
  k_packW<<<dim3((K1 * PROJ) / 256), 256, 0, stream>>>(Wx_w, Wtp_w, W1p);
  k_efv<<<dim3(1), 512, 0, stream>>>(f_idx, emb_f, fea_w, fea_b, Efv);
  k_wk<<<dim3(6), 256, 0, stream>>>(conv_w, Efv, Wk);

  k_gemm1<<<dim3(MT1 / 8, PROJ / 64), 256, 0, stream>>>(Apack, W1p, Wx_b, Wtp_b, ExPack);

  k_exm<<<dim3(Bn), 512, 0, stream>>>(ExPack, vseq_w, vseq_b, Exm);
  k_stats<<<dim3(Bn), 256, 0, stream>>>(Exm, Wk, conv_b, Msm, invZ);

  k_gemm2<<<dim3(Bn, PROJ / 64), 256, 0, stream>>>(ExPack, Exm, Wk, conv_b, Msm, invZ, ExOut);

  k_visit<<<dim3(Bn), 256, 0, stream>>>(ExOut, visit_w, visit_b, xbase, sproj_w, sproj_b, ExAll);
  k_adj<<<dim3(Bn), 256, 0, stream>>>(ExAll, phi, adj);

  // GCN: h = relu(adj @ (ExAll@gc1_w) + gc1_b); h2 = adj @ (h@gc2_w) + gc2_b
  k_mm<<<dim3(Bn, 1), 256, Dn  * 4, stream>>>(ExAll, gc1_w, nullptr, t1,   Dn,  H1n, 0);
  k_mm<<<dim3(Bn, 1), 256, Bn  * 4, stream>>>(adj,   t1,    gc1_b,  hbuf, Bn,  H1n, 1);
  k_mm<<<dim3(Bn, 1), 256, H1n * 4, stream>>>(hbuf,  gc2_w, nullptr, t2,   H1n, H2n, 0);
  k_mm<<<dim3(Bn, 1), 256, Bn  * 4, stream>>>(adj,   t2,    gc2_b,  xoth, Bn,  H2n, 0);
  k_lsm<<<dim3(Bn), 128, 0, stream>>>(xoth, xoth);

  // xo = x_other @ agg_w + agg_b
  k_mm<<<dim3(Bn, (Dn + 255) / 256), 256, H2n * 4, stream>>>(xoth, agg_w, agg_b, xo, H2n, Dn, 0);

  k_final<<<dim3(Bn), 256, 0, stream>>>(ExAll, xo, agg1_w, agg1_b, agg2_w, agg2_b, wy_w, wy_b, outp);
}
